// EGNN_Layer_31602369364715
// MI455X (gfx1250) — matrix operand order repaired and checked by reading, not compiled, not run on hardware
//
#include <hip/hip_runtime.h>

#define Hdim 128
#define EDdim 16
#define KPAD 288   // 273 padded to 9 chunks of K=32

typedef __attribute__((ext_vector_type(16))) __bf16 v16bf;
typedef __attribute__((ext_vector_type(8)))  float  v8f;
typedef __attribute__((ext_vector_type(8)))  unsigned short us8;

__device__ __forceinline__ unsigned short f2bf_u(float f) {
  union { float f; unsigned u; } v; v.f = f;
  unsigned r = v.u + 0x7FFFu + ((v.u >> 16) & 1u);   // round-to-nearest-even
  return (unsigned short)(r >> 16);
}

__device__ __forceinline__ float silu(float x) { return x / (1.0f + __expf(-x)); }

__device__ __forceinline__ v8f wmma_bf16(v16bf a, v16bf b, v8f c) {
  return __builtin_amdgcn_wmma_f32_16x16x32_bf16(false, a, false, b, (short)0, c, false, false);
}

// A fragment: 16x32 bf16 tile from a row-major bf16(u16) buffer (lda elems), chunk at k0.
// ISA layout: lanes 0-15 (M=lane): K = 0..7 (v0-3) and 16..23 (v4-7); lanes 16-31: +8.
// -> per lane: two contiguous 16B loads.
__device__ __forceinline__ v16bf load_a_frag(const unsigned short* A, int lda, int k0, int lane) {
  const int g = (lane >> 4) & 1, m = lane & 15;
  union { us8 h[2]; v16bf v; } u;
  u.h[0] = *(const us8*)(A + m * lda + k0 + 8 * g);
  u.h[1] = *(const us8*)(A + m * lda + k0 + 16 + 8 * g);
  return u.v;
}

// B fragment from pre-packed weights: frag (kt,nt) stored as 32 lanes x 16 contiguous bf16.
__device__ __forceinline__ v16bf load_b_frag(const unsigned short* Wp, int kt, int nt, int lane) {
  return *(const v16bf*)(Wp + (((kt * 8 + nt) * 32) + lane) * 16);
}

// ---------------- weight packing: fp32 (K x 128) -> bf16 WMMA B-fragment order --------
// packed index p = ((kt*8+nt)*32 + lane)*16 + i ;  k = kt*32 + 8*(lane>>4) + (i<8 ? i : 8+i) ; n = nt*16 + (lane&15)
#define EW1P 0
#define EW2P 36864
#define CW1P 53248
#define NW1P 69632
#define NW2P 102400
#define WPTOT 118784

__global__ void egnn_pack_weights(const float* __restrict__ ew1, const float* __restrict__ ew2,
                                  const float* __restrict__ cw1, const float* __restrict__ nw1,
                                  const float* __restrict__ nw2, unsigned short* __restrict__ wp) {
  int i = blockIdx.x * blockDim.x + threadIdx.x;
  if (i >= WPTOT) return;
  const float* W; int Kv, base, p;
  if      (i < EW2P) { W = ew1; Kv = 273; base = EW1P; p = i - EW1P; }
  else if (i < CW1P) { W = ew2; Kv = 128; base = EW2P; p = i - EW2P; }
  else if (i < NW1P) { W = cw1; Kv = 128; base = CW1P; p = i - CW1P; }
  else if (i < NW2P) { W = nw1; Kv = 256; base = NW1P; p = i - NW1P; }
  else               { W = nw2; Kv = 128; base = NW2P; p = i - NW2P; }
  int ii = p & 15, lane = (p >> 4) & 31, fr = p >> 9;
  int nt = fr & 7, kt = fr >> 3;
  int k = kt * 32 + 8 * (lane >> 4) + (ii < 8 ? ii : 8 + ii);
  int n = nt * 16 + (lane & 15);
  wp[base + p] = (k < Kv) ? f2bf_u(W[k * Hdim + n]) : (unsigned short)0;
}

__global__ void egnn_init(const float* __restrict__ x, float* __restrict__ m_i,
                          float* __restrict__ x_out, int N) {
  int i = blockIdx.x * blockDim.x + threadIdx.x;
  if (i < N * Hdim) m_i[i] = 0.0f;
  if (i < N * 3)    x_out[i] = x[i];
}

// ---------------- edge kernel: one wave = 16 edges -----------------------------------
__global__ __launch_bounds__(128) void egnn_edge(
    const float* __restrict__ h, const float* __restrict__ x,
    const long long* __restrict__ eidx, const float* __restrict__ eattr,
    const unsigned short* __restrict__ ew1p, const float* __restrict__ eb1,
    const unsigned short* __restrict__ ew2p, const float* __restrict__ eb2,
    const float* __restrict__ aw, const float* __restrict__ ab,
    const unsigned short* __restrict__ cw1p, const float* __restrict__ cb1,
    const float* __restrict__ cw2,
    float* __restrict__ m_i, float* __restrict__ x_out, int E, int N) {
  __shared__ __align__(16) unsigned char smem[4 * 13696];
  const int lane = threadIdx.x & 31;
  const int wave = threadIdx.x >> 5;
  const int e0 = (blockIdx.x * 4 + wave) * 16;
  if (e0 >= E) return;

  unsigned char* wb = smem + wave * 13696;
  unsigned short* A = (unsigned short*)wb;           // 16x288 bf16 (9216B); reused as F f32 16x128
  unsigned short* M = (unsigned short*)(wb + 9216);  // 16x128 bf16
  int*   rowS = (int*)  (wb + 13312);
  float* radS = (float*)(wb + 13376);
  float* cdS  = (float*)(wb + 13440);                // 16x3
  float* attS = (float*)(wb + 13632);
  float* F    = (float*)A;

  const int g = lane >> 4;
  const int e = lane & 15;
  const int eg = min(e0 + e, E - 1);
  const bool ev = (e0 + e) < E;

  const long long r = eidx[eg];
  const long long c = eidx[(long long)E + eg];

  // Build A tile: [h[row](128) | h[col](128) | radial(1) | edge_attr(16) | pad(15)]
  if (g == 0) {
    float dx = x[r * 3 + 0] - x[c * 3 + 0];
    float dy = x[r * 3 + 1] - x[c * 3 + 1];
    float dz = x[r * 3 + 2] - x[c * 3 + 2];
    float rad = dx * dx + dy * dy + dz * dz;
    rowS[e] = (int)r; radS[e] = rad;
    cdS[e * 3 + 0] = dx; cdS[e * 3 + 1] = dy; cdS[e * 3 + 2] = dz;
    A[e * KPAD + 256] = f2bf_u(rad);
    #pragma unroll
    for (int j = 0; j < 8; ++j) A[e * KPAD + 257 + j] = f2bf_u(eattr[(long long)eg * EDdim + j]);
    #pragma unroll
    for (int j = 273; j < 280; ++j) A[e * KPAD + j] = 0;
  } else {
    #pragma unroll
    for (int j = 8; j < 16; ++j) A[e * KPAD + 257 + j] = f2bf_u(eattr[(long long)eg * EDdim + j]);
    #pragma unroll
    for (int j = 280; j < 288; ++j) A[e * KPAD + j] = 0;
  }
  const float* hr = h + r * Hdim;
  const float* hc = h + c * Hdim;
  const int f0 = g * 64;
  for (int f = f0; f < f0 + 64; ++f) {
    A[e * KPAD + f]       = f2bf_u(hr[f]);
    A[e * KPAD + 128 + f] = f2bf_u(hc[f]);
  }
  __builtin_amdgcn_wave_barrier();

  // GEMM1: m = silu(A @ ew1 + eb1) -> M (bf16)
  for (int n = 0; n < 8; ++n) {
    v8f acc = {};
    #pragma unroll
    for (int k = 0; k < 9; ++k)
      acc = wmma_bf16(load_a_frag(A, KPAD, k * 32, lane), load_b_frag(ew1p, k, n, lane), acc);
    const int nc = n * 16 + e;
    const float bias = eb1[nc];
    #pragma unroll
    for (int v = 0; v < 8; ++v)
      M[(v + 8 * g) * Hdim + nc] = f2bf_u(silu(acc[v] + bias));
  }
  __builtin_amdgcn_wave_barrier();

  // GEMM2: m_ij = silu(M @ ew2 + eb2) -> F (f32, reuses A space)
  for (int n = 0; n < 8; ++n) {
    v8f acc = {};
    #pragma unroll
    for (int k = 0; k < 4; ++k)
      acc = wmma_bf16(load_a_frag(M, Hdim, k * 32, lane), load_b_frag(ew2p, k, n, lane), acc);
    const int nc = n * 16 + e;
    const float bias = eb2[nc];
    #pragma unroll
    for (int v = 0; v < 8; ++v)
      F[(v + 8 * g) * Hdim + nc] = silu(acc[v] + bias);
  }
  __builtin_amdgcn_wave_barrier();

  // attention head + m_att; scatter-add into m_i; stage m_att (bf16) in M for GEMM3
  if (g == 0) {
    float s = ab[0];
    for (int f = 0; f < Hdim; ++f) s += F[e * Hdim + f] * aw[f];
    attS[e] = 1.0f / (1.0f + __expf(-s));
  }
  __builtin_amdgcn_wave_barrier();
  {
    const float at = attS[e];
    const long long rbase = (long long)rowS[e] * Hdim;
    for (int f = f0; f < f0 + 64; ++f) {
      float v = F[e * Hdim + f] * at;
      M[e * Hdim + f] = f2bf_u(v);
      if (ev) atomicAdd(&m_i[rbase + f], v);
    }
  }
  __builtin_amdgcn_wave_barrier();

  // GEMM3: t = silu(m_att @ cw1 + cb1) -> F
  for (int n = 0; n < 8; ++n) {
    v8f acc = {};
    #pragma unroll
    for (int k = 0; k < 4; ++k)
      acc = wmma_bf16(load_a_frag(M, Hdim, k * 32, lane), load_b_frag(cw1p, k, n, lane), acc);
    const int nc = n * 16 + e;
    const float bias = cb1[nc];
    #pragma unroll
    for (int v = 0; v < 8; ++v)
      F[(v + 8 * g) * Hdim + nc] = silu(acc[v] + bias);
  }
  __builtin_amdgcn_wave_barrier();

  // coord head: coord_w = t @ cw2 ; scatter coord updates
  if (g == 0 && ev) {
    float s = 0.0f;
    for (int f = 0; f < Hdim; ++f) s += F[e * Hdim + f] * cw2[f];
    const float cwv = s * rsqrtf(radS[e] + 1e-8f);
    const int rr = rowS[e];
    atomicAdd(&x_out[rr * 3 + 0], cdS[e * 3 + 0] * cwv);
    atomicAdd(&x_out[rr * 3 + 1], cdS[e * 3 + 1] * cwv);
    atomicAdd(&x_out[rr * 3 + 2], cdS[e * 3 + 2] * cwv);
  }
}

// ---------------- node kernel: one wave = 16 nodes -----------------------------------
__global__ __launch_bounds__(128) void egnn_node(
    const float* __restrict__ h, const float* __restrict__ m_i,
    const unsigned short* __restrict__ nw1p, const float* __restrict__ nb1,
    const unsigned short* __restrict__ nw2p, const float* __restrict__ nb2,
    float* __restrict__ h_out, int N) {
  __shared__ __align__(16) unsigned char smem[4 * 12288];
  const int lane = threadIdx.x & 31;
  const int wave = threadIdx.x >> 5;
  const int n0 = (blockIdx.x * 4 + wave) * 16;
  if (n0 >= N) return;

  unsigned char* wb = smem + wave * 12288;
  unsigned short* A = (unsigned short*)wb;           // 16x256 bf16
  unsigned short* T = (unsigned short*)(wb + 8192);  // 16x128 bf16

  const int g = lane >> 4;
  const int m = lane & 15;
  const int node = min(n0 + m, N - 1);
  const int f0 = g * 64;
  for (int f = f0; f < f0 + 64; ++f) {
    A[m * 256 + f]       = f2bf_u(h[(long long)node * Hdim + f]);
    A[m * 256 + 128 + f] = f2bf_u(m_i[(long long)node * Hdim + f]);
  }
  __builtin_amdgcn_wave_barrier();

  // GEMM1: t = silu([h|m_i] @ nw1 + nb1)
  for (int n = 0; n < 8; ++n) {
    v8f acc = {};
    #pragma unroll
    for (int k = 0; k < 8; ++k)
      acc = wmma_bf16(load_a_frag(A, 256, k * 32, lane), load_b_frag(nw1p, k, n, lane), acc);
    const int nc = n * 16 + m;
    const float bias = nb1[nc];
    #pragma unroll
    for (int v = 0; v < 8; ++v)
      T[(v + 8 * g) * Hdim + nc] = f2bf_u(silu(acc[v] + bias));
  }
  __builtin_amdgcn_wave_barrier();

  // GEMM2: h_out = h + (t @ nw2 + nb2)
  for (int n = 0; n < 8; ++n) {
    v8f acc = {};
    #pragma unroll
    for (int k = 0; k < 4; ++k)
      acc = wmma_bf16(load_a_frag(T, Hdim, k * 32, lane), load_b_frag(nw2p, k, n, lane), acc);
    const int nc = n * 16 + m;
    const float bias = nb2[nc];
    #pragma unroll
    for (int v = 0; v < 8; ++v) {
      const int node2 = n0 + v + 8 * g;
      if (node2 < N)
        h_out[(long long)node2 * Hdim + nc] = h[(long long)node2 * Hdim + nc] + acc[v] + bias;
    }
  }
}

extern "C" void kernel_launch(void* const* d_in, const int* in_sizes, int n_in,
                              void* d_out, int out_size, void* d_ws, size_t ws_size,
                              hipStream_t stream) {
  const float*     h     = (const float*)d_in[0];
  const float*     x     = (const float*)d_in[1];
  const long long* eidx  = (const long long*)d_in[2];
  const float*     eattr = (const float*)d_in[3];
  const float*     ew1   = (const float*)d_in[4];
  const float*     eb1   = (const float*)d_in[5];
  const float*     ew2   = (const float*)d_in[6];
  const float*     eb2   = (const float*)d_in[7];
  const float*     aw    = (const float*)d_in[8];
  const float*     ab    = (const float*)d_in[9];
  const float*     cw1   = (const float*)d_in[10];
  const float*     cb1   = (const float*)d_in[11];
  const float*     cw2   = (const float*)d_in[12];
  const float*     nw1   = (const float*)d_in[13];
  const float*     nb1   = (const float*)d_in[14];
  const float*     nw2   = (const float*)d_in[15];
  const float*     nb2   = (const float*)d_in[16];

  const int N = in_sizes[0] / Hdim;
  const int E = in_sizes[2] / 2;

  unsigned short* wp  = (unsigned short*)d_ws;
  float* m_i   = (float*)((char*)d_ws + (size_t)WPTOT * 2);   // 237568 B, 256-aligned
  float* h_out = (float*)d_out;
  float* x_out = h_out + (size_t)N * Hdim;

  egnn_pack_weights<<<(WPTOT + 255) / 256, 256, 0, stream>>>(ew1, ew2, cw1, nw1, nw2, wp);
  egnn_init<<<(N * Hdim + 255) / 256, 256, 0, stream>>>(x, m_i, x_out, N);

  const int etiles = (E + 15) / 16;
  egnn_edge<<<(etiles + 3) / 4, 128, 0, stream>>>(h, x, eidx, eattr,
      wp + EW1P, eb1, wp + EW2P, eb2, aw, ab, wp + CW1P, cb1, cw2, m_i, x_out, E, N);

  const int ntiles = (N + 15) / 16;
  egnn_node<<<(ntiles + 3) / 4, 128, 0, stream>>>(h, m_i, wp + NW1P, nb1, wp + NW2P, nb2, h_out, N);
}
